// Conv3DRecurrentInhibition_45483703665094
// MI455X (gfx1250) — matrix operand order repaired and checked by reading, not compiled, not run on hardware
//
#include <hip/hip_runtime.h>

// ---------------------------------------------------------------------------
// s10 = (sum_{k=0}^{10} C^k) x  where C = banded (15-tap Ricker) conv along
// the 256-channel axis with zero boundary. We build M = sum C^k exactly
// (boundary rows differ from the Toeplitz interior!) then do one f16-WMMA
// GEMM: out[b] = M (256x256) @ X[b] (256x3136).  HBM-bound at ~0.21 GB.
// ---------------------------------------------------------------------------

typedef __attribute__((ext_vector_type(16))) _Float16     v16h;
typedef __attribute__((ext_vector_type(8)))  float        v8f;
typedef __attribute__((ext_vector_type(4)))  unsigned int u32x4;
typedef __attribute__((ext_vector_type(4)))  float        f32x4;

#define D_CH    256
#define HW      3136
#define BSTRIDE (D_CH * HW)      // 802816 elements per batch
#define NBATCH  32
#define SCOPE_  15
#define PAD_    7
#define NSTEPS  10

#define BM 128
#define BN 64
#define BK 32
#define LDA 40                   // padded LDS row stride (f16) -> conflict-free b128 frag loads
#define LDB 40

// ---- Phase 1: build M = sum_{k=0}^{10} C^k ---------------------------------

__global__ void op_init_identity(float* __restrict__ A) {
  const int d = blockIdx.x, j = threadIdx.x;
  A[d * D_CH + j] = (d == j) ? 1.0f : 0.0f;
}

// A_next = I + C * A_prev   (C = 15-tap conv along rows, zero boundary)
__global__ void op_iterate(float* __restrict__ dst, const float* __restrict__ src,
                           const float* __restrict__ w) {
  const int d = blockIdx.x, j = threadIdx.x;
  float acc = (d == j) ? 1.0f : 0.0f;
#pragma unroll
  for (int t = 0; t < SCOPE_; ++t) {
    const int r = d - PAD_ + t;
    if (r >= 0 && r < D_CH) acc = fmaf(w[t], src[r * D_CH + j], acc);
  }
  dst[d * D_CH + j] = acc;
}

__global__ void op_convert_f16(_Float16* __restrict__ Mh, const float* __restrict__ Af) {
  const int i = blockIdx.x * blockDim.x + threadIdx.x;
  Mh[i] = (_Float16)Af[i];
}

// ---- Phase 2: out[b] = M @ X[b] via v_wmma_f32_16x16x32_f16 ----------------

__global__ __launch_bounds__(256) void op_gemm(float* __restrict__ out,
                                               const float* __restrict__ X,
                                               const _Float16* __restrict__ Mh) {
  __shared__ _Float16 As[BM * LDA];   // 10240 B: M tile, row-major [m][k]
  __shared__ _Float16 Bs[BN * LDB];   //  5120 B: X tile, transposed [n][k]

  const int tid  = threadIdx.x;
  const int wave = tid >> 5;
  const int lane = tid & 31;
  const int half = lane >> 4;
  const int l16  = lane & 15;
  const int wm   = wave & 3;          // 4 waves along M
  const int wn   = wave >> 2;         // 2 waves along N

  const int n0 = blockIdx.x * BN;     // 49 tiles cover HW=3136 exactly
  const int m0 = blockIdx.y * BM;     // 2 tiles cover 256 rows exactly
  const int b  = blockIdx.z;

  const float* Xb = X + (size_t)b * BSTRIDE;

  v8f acc[2][2] = {};                 // wave tile 32x32 = 2x2 of 16x16

  const int ar  = tid >> 1;           // A staging: 128 rows x 32 cols
  const int ac  = (tid & 1) * 16;
  const int bnl = (tid & 15) * 4;     // B staging: 2(k) x 4(n) micro-tile/thread
  const int bkl = (tid >> 4) * 2;     //   lanes 0-15 read 64 consecutive cols (coalesced)

  for (int kk = 0; kk < D_CH / BK; ++kk) {
    const int k0 = kk * BK;

    { // stage A tile (f16, 16B-aligned vector copies)
      const u32x4* g = (const u32x4*)(Mh + (m0 + ar) * D_CH + k0 + ac);
      u32x4* s = (u32x4*)(&As[ar * LDA + ac]);
      s[0] = g[0];
      s[1] = g[1];
    }
    { // stage B tile: coalesced f32 loads of rows k, k+1; convert -> f16,
      // pack the (k,k+1) pair per column and store transposed as one b32
      const f32x4* g0 = (const f32x4*)(Xb + (k0 + bkl    ) * HW + n0 + bnl);
      const f32x4* g1 = (const f32x4*)(Xb + (k0 + bkl + 1) * HW + n0 + bnl);
      const f32x4 f0 = g0[0];
      const f32x4 f1 = g1[0];
#pragma unroll
      for (int i = 0; i < 4; ++i) {
        union { _Float16 h[2]; unsigned int u; } p;
        p.h[0] = (_Float16)f0[i];
        p.h[1] = (_Float16)f1[i];
        *(unsigned int*)(&Bs[(bnl + i) * LDB + bkl]) = p.u;
      }
    }
    __syncthreads();

    union Frag { u32x4 q[2]; v16h v; };
    Frag a[2], bf;
#pragma unroll
    for (int mi = 0; mi < 2; ++mi) {  // A frag: lane m=l16; K chunks at half*8, 16+half*8
      const int mr = wm * 32 + mi * 16 + l16;
      a[mi].q[0] = *(const u32x4*)(&As[mr * LDA + half * 8]);
      a[mi].q[1] = *(const u32x4*)(&As[mr * LDA + 16 + half * 8]);
    }
#pragma unroll
    for (int ni = 0; ni < 2; ++ni) {  // B frag: lane n=l16; K chunks at half*8, 16+half*8
      const int nr = wn * 32 + ni * 16 + l16;
      bf.q[0] = *(const u32x4*)(&Bs[nr * LDB + half * 8]);
      bf.q[1] = *(const u32x4*)(&Bs[nr * LDB + 16 + half * 8]);
#pragma unroll
      for (int mi = 0; mi < 2; ++mi) {
        acc[mi][ni] = __builtin_amdgcn_wmma_f32_16x16x32_f16(
            false, a[mi].v, false, bf.v, (short)0, acc[mi][ni], false, false);
      }
    }
    __syncthreads();
  }

  // epilogue: D layout — VGPR r -> row m = 8*half + r, col = lane%16.
  // Non-temporal stores: output is write-once streaming; keep X/M in L2.
  float* outB = out + (size_t)b * BSTRIDE;
#pragma unroll
  for (int mi = 0; mi < 2; ++mi) {
#pragma unroll
    for (int ni = 0; ni < 2; ++ni) {
      const int col = n0 + wn * 32 + ni * 16 + l16;
#pragma unroll
      for (int r = 0; r < 8; ++r) {
        const int row = m0 + wm * 32 + mi * 16 + half * 8 + r;
        __builtin_nontemporal_store(acc[mi][ni][r], &outB[row * HW + col]);
      }
    }
  }
}

// ---------------------------------------------------------------------------

extern "C" void kernel_launch(void* const* d_in, const int* in_sizes, int n_in,
                              void* d_out, int out_size, void* d_ws, size_t ws_size,
                              hipStream_t stream) {
  (void)in_sizes; (void)n_in; (void)out_size; (void)ws_size;

  const float* X = (const float*)d_in[0];   // (32,256,56,56) fp32
  const float* w = (const float*)d_in[1];   // (15,) fp32
  float* out = (float*)d_out;

  // workspace: A0 (256KB f32) | A1 (256KB f32) | Mh (128KB f16)  = 640 KB
  float*    A0 = (float*)d_ws;
  float*    A1 = A0 + D_CH * D_CH;
  _Float16* Mh = (_Float16*)(A1 + D_CH * D_CH);

  op_init_identity<<<D_CH, D_CH, 0, stream>>>(A0);
  float* src = A0;
  float* dst = A1;
  for (int s = 0; s < NSTEPS; ++s) {        // MAX_STEPS = 10 (fixed in reference)
    op_iterate<<<D_CH, D_CH, 0, stream>>>(dst, src, w);
    float* t = src; src = dst; dst = t;
  }
  op_convert_f16<<<D_CH, D_CH, 0, stream>>>(Mh, src);

  op_gemm<<<dim3(HW / BN, D_CH / BM, NBATCH), 256, 0, stream>>>(out, X, Mh);
}
